// Attention_64647847739871
// MI455X (gfx1250) — matrix operand order repaired and checked by reading, not compiled, not run on hardware
//
#include <hip/hip_runtime.h>
#include <hip/hip_bf16.h>

typedef __attribute__((ext_vector_type(16))) _Float16 v16h;
typedef __attribute__((ext_vector_type(8)))  float    v8f;
typedef int v4i_ __attribute__((vector_size(16)));

#define B_    16
#define N_    1024
#define H_    16
#define D_    128
#define DIM_  2048
#define DH_   2048
#define QKVO_ 6144
#define NOFF_ 1024
#define SCALE_ 0.08838834764831845f   /* 128^-0.5 */

// CDNA5 async global->LDS path (ASYNCcnt), guarded so compile never breaks.
#if defined(__has_builtin)
#if __has_builtin(__builtin_amdgcn_global_load_async_to_lds_b128) && \
    __has_builtin(__builtin_amdgcn_s_wait_asynccnt)
#define HAVE_ASYNC_LDS 1
#endif
#endif

__device__ __forceinline__ void copy16_g2lds(unsigned int* lds, const unsigned int* g) {
#if HAVE_ASYNC_LDS
  __builtin_amdgcn_global_load_async_to_lds_b128(
      (__attribute__((address_space(1))) v4i_*)(const void*)g,
      (__attribute__((address_space(3))) v4i_*)(void*)lds, 0, 0);
#else
  *(uint4*)lds = *(const uint4*)g;
#endif
}

__device__ __forceinline__ void wait_async_lds() {
#if HAVE_ASYNC_LDS
  __builtin_amdgcn_s_wait_asynccnt(0);
#endif
}

// ---------------------------------------------------------------------------
// Kernel 1: LayerNorm (f32 in) -> xn (f16), one row (2048) per 256-thread block
// ---------------------------------------------------------------------------
__global__ __launch_bounds__(256) void ln_kernel(
    const float* __restrict__ x, const float* __restrict__ g,
    const float* __restrict__ bta, _Float16* __restrict__ xn) {
  int row = blockIdx.x;                       // 0..16383
  const float* xr = x + (size_t)row * DIM_;
  float s = 0.f, s2 = 0.f;
  for (int i = threadIdx.x; i < DIM_; i += 256) { float v = xr[i]; s += v; s2 += v * v; }
  __shared__ float red0[256], red1[256];
  red0[threadIdx.x] = s; red1[threadIdx.x] = s2;
  __syncthreads();
  for (int off = 128; off > 0; off >>= 1) {
    if (threadIdx.x < off) {
      red0[threadIdx.x] += red0[threadIdx.x + off];
      red1[threadIdx.x] += red1[threadIdx.x + off];
    }
    __syncthreads();
  }
  float mean = red0[0] * (1.0f / DIM_);
  float var  = red1[0] * (1.0f / DIM_) - mean * mean;
  float rstd = rsqrtf(var + 1e-5f);
  _Float16* xo = xn + (size_t)row * DIM_;
  for (int i = threadIdx.x; i < DIM_; i += 256)
    xo[i] = (_Float16)(((xr[i] - mean) * rstd) * g[i] + bta[i]);
}

// ---------------------------------------------------------------------------
// Kernel 2: weight convert+transpose: w (K x Ncol, f32) -> wt (Ncol x K, f16)
// ---------------------------------------------------------------------------
__global__ __launch_bounds__(256) void wt_kernel(
    const float* __restrict__ w, _Float16* __restrict__ wt, int K, int Ncol) {
  size_t i = (size_t)blockIdx.x * 256 + threadIdx.x;
  if (i >= (size_t)K * Ncol) return;
  int n = (int)(i % Ncol);
  int k = (int)(i / Ncol);
  wt[(size_t)n * K + k] = (_Float16)w[i];
}

// ---------------------------------------------------------------------------
// Kernel 3: tiled WMMA GEMM  C(MxN) = A(MxK,f16) * Bt(NxK,f16)^T + bias
//  workgroup tile 128x64, 8 waves (4x2), each wave 32x32, K step 32.
//  Double-buffered LDS: async-load tile t+1 while WMMAs run on tile t;
//  one s_wait_asynccnt + one barrier per K-step.
//  mode 0: scatter into q/k/vT (f16) with qkv_b.  mode 1: f32 out + proj_b.
// ---------------------------------------------------------------------------
__global__ __launch_bounds__(256) void gemm_kernel(
    const _Float16* __restrict__ A, const _Float16* __restrict__ Bt,
    const float* __restrict__ bias, int K, int mode,
    _Float16* __restrict__ q, _Float16* __restrict__ k,
    _Float16* __restrict__ vt, float* __restrict__ out) {
  __shared__ __align__(16) unsigned int Atile[2][128 * 20];  // pitch 20 u32
  __shared__ __align__(16) unsigned int Btile[2][64 * 20];
  const unsigned int* A32 = (const unsigned int*)A;
  const unsigned int* B32 = (const unsigned int*)Bt;
  int m0 = blockIdx.y * 128;
  int n0 = blockIdx.x * 64;
  int tid = threadIdx.x;
  int wid = tid >> 5, lane = tid & 31, half = lane >> 4, l16 = lane & 15;
  int wm = wid >> 1, wn = wid & 1;
  int Ku = K >> 1;                           // u32 per row

  auto stage = [&](int buf, int kk) {
    int rB = tid >> 2, cB = (tid & 3) << 2;
    copy16_g2lds(&Btile[buf][rB * 20 + cB],
                 &B32[(size_t)(n0 + rB) * Ku + (kk >> 1) + cB]);
#pragma unroll
    for (int it = 0; it < 2; ++it) {
      int idx = it * 256 + tid;
      int rA = idx >> 2, cA = (idx & 3) << 2;
      copy16_g2lds(&Atile[buf][rA * 20 + cA],
                   &A32[(size_t)(m0 + rA) * Ku + (kk >> 1) + cA]);
    }
  };

  v8f zero = {};
  v8f acc00 = zero, acc01 = zero, acc10 = zero, acc11 = zero;

  stage(0, 0);
  wait_async_lds();
  __syncthreads();

  for (int k0 = 0; k0 < K; k0 += 32) {
    int cur = (k0 >> 5) & 1;
    if (k0 + 32 < K) stage(1 - cur, k0 + 32);   // overlap with WMMAs below

    // A fragment: row l16; u32 offsets {0..3}+half*4 and {8..11}+half*4 -> 2x b128
    // B fragment: col l16; u32 offsets {0..3}+half*4 and {8..11}+half*4 -> 2x b128
    union U16 { v16h h; unsigned int u[8]; uint4 v4[2]; };
    U16 af[2], bf[2];
#pragma unroll
    for (int t = 0; t < 2; ++t) {
      const unsigned int* ra = &Atile[cur][(wm * 32 + t * 16 + l16) * 20 + half * 4];
      const unsigned int* rb = &Btile[cur][(wn * 32 + t * 16 + l16) * 20 + half * 4];
      af[t].v4[0] = *(const uint4*)(ra);
      af[t].v4[1] = *(const uint4*)(ra + 8);
      bf[t].v4[0] = *(const uint4*)(rb);
      bf[t].v4[1] = *(const uint4*)(rb + 8);
    }
    acc00 = __builtin_amdgcn_wmma_f32_16x16x32_f16(false, af[0].h, false, bf[0].h,
                                                   (short)0, acc00, false, false);
    acc01 = __builtin_amdgcn_wmma_f32_16x16x32_f16(false, af[0].h, false, bf[1].h,
                                                   (short)0, acc01, true, false);
    acc10 = __builtin_amdgcn_wmma_f32_16x16x32_f16(false, af[1].h, false, bf[0].h,
                                                   (short)0, acc10, false, false);
    acc11 = __builtin_amdgcn_wmma_f32_16x16x32_f16(false, af[1].h, false, bf[1].h,
                                                   (short)0, acc11, true, false);

    // next tile's async writes done + all waves done reading tile `cur`
    wait_async_lds();
    __syncthreads();
  }

  // epilogue (C layout: lane holds M = r + 8*half, N = l16)
  v8f accs[2][2] = {{acc00, acc01}, {acc10, acc11}};
#pragma unroll
  for (int i = 0; i < 2; ++i)
#pragma unroll
    for (int j = 0; j < 2; ++j)
#pragma unroll
      for (int r = 0; r < 8; ++r) {
        int m = m0 + wm * 32 + i * 16 + r + half * 8;
        int n = n0 + wn * 32 + j * 16 + l16;
        float val = accs[i][j][r] + bias[n];
        if (mode == 0) {
          int b = m >> 10, nn = m & 1023;
          int which = n >> 11, rem = n & 2047;
          int h = rem >> 7, d = rem & 127;
          size_t bh = (size_t)(b * H_ + h);
          if (which == 0)
            q[bh * N_ * D_ + (size_t)nn * D_ + d] = (_Float16)val;
          else if (which == 1)
            k[bh * N_ * D_ + (size_t)nn * D_ + d] = (_Float16)val;
          else
            vt[bh * D_ * N_ + (size_t)d * N_ + nn] = (_Float16)val;
        } else {
          out[(size_t)m * DIM_ + n] = val;
        }
      }
}

// ---------------------------------------------------------------------------
// Kernel 4: flash attention. block = 4 waves; wave owns 16 query rows.
//  Streams 32 key columns/step: S = Q K^T (8 WMMA), bias gather, online
//  softmax, P->LDS->A-fragment, O += P V (8 WMMA). O kept in registers.
//  All fragment loads are b128 (uint4).
// ---------------------------------------------------------------------------
__global__ __launch_bounds__(128) void attn_kernel(
    const _Float16* __restrict__ qb, const _Float16* __restrict__ kb,
    const _Float16* __restrict__ vtb, const float* __restrict__ abias,
    const int* __restrict__ idxs, _Float16* __restrict__ ob) {
  __shared__ float bcache[NOFF_];
  __shared__ __align__(16) _Float16 Pl[4][16 * 40];   // per-wave 16x32 (pitch 40 f16)
  int bh = blockIdx.x >> 4;                 // (b*H + h)
  int qt = blockIdx.x & 15;
  int b = bh >> 4, h = bh & 15;
  for (int i = threadIdx.x; i < NOFF_; i += 128) bcache[i] = abias[h * NOFF_ + i];
  __syncthreads();

  int wid = threadIdx.x >> 5, lane = threadIdx.x & 31;
  int half = lane >> 4, l16 = lane & 15;
  int q0 = qt * 64 + wid * 16;
  const unsigned int* q32 = (const unsigned int*)qb;
  const unsigned int* k32 = (const unsigned int*)kb;
  const unsigned int* v32 = (const unsigned int*)vtb;
  union U16 { v16h h; unsigned int u[8]; uint4 v4[2]; };

  // Q fragments: 4 x (16x32), A layout; per frag two b128 loads
  U16 qf[4];
  size_t qrow = ((size_t)bh * N_ + q0 + l16) * 64;   // 64 u32 per q row
#pragma unroll
  for (int f = 0; f < 4; ++f) {
    const unsigned int* p = &q32[qrow + f * 16 + half * 4];
    qf[f].v4[0] = *(const uint4*)(p);
    qf[f].v4[1] = *(const uint4*)(p + 8);
  }

  float mrun[8], lrun[8];
  v8f zero = {};
  v8f Oacc[8];
#pragma unroll
  for (int r = 0; r < 8; ++r) { mrun[r] = -1e30f; lrun[r] = 0.f; }
#pragma unroll
  for (int dt = 0; dt < 8; ++dt) Oacc[dt] = zero;

  for (int j0 = 0; j0 < N_; j0 += 32) {
    // K fragments (B layout): rows j0+l16 / j0+16+l16, u32 {0..3}+half*4 and {8..11}+half*4 per frag
    U16 kf0[4], kf1[4];
    size_t kr0 = ((size_t)bh * N_ + j0 + l16) * 64;
    size_t kr1 = kr0 + 16 * 64;
    if (j0 + 32 < N_) __builtin_prefetch(&k32[kr0 + 32 * 64], 0, 0);
#pragma unroll
    for (int f = 0; f < 4; ++f) {
      const unsigned int* p0p = &k32[kr0 + f * 16 + half * 4];
      const unsigned int* p1p = &k32[kr1 + f * 16 + half * 4];
      kf0[f].v4[0] = *(const uint4*)(p0p);
      kf0[f].v4[1] = *(const uint4*)(p0p + 8);
      kf1[f].v4[0] = *(const uint4*)(p1p);
      kf1[f].v4[1] = *(const uint4*)(p1p + 8);
    }
    v8f s0 = zero, s1 = zero;
#pragma unroll
    for (int f = 0; f < 4; ++f) {
      s0 = __builtin_amdgcn_wmma_f32_16x16x32_f16(false, qf[f].h, false, kf0[f].h,
                                                  (short)0, s0, false, false);
      s1 = __builtin_amdgcn_wmma_f32_16x16x32_f16(false, qf[f].h, false, kf1[f].h,
                                                  (short)0, s1, true, false);
    }

    // bias gather + scale + row max (rows live across 16-lane halves)
    float p0[8], p1[8], rm[8];
#pragma unroll
    for (int r = 0; r < 8; ++r) {
      int row = q0 + r + half * 8;
      int c0 = j0 + l16;
      float a0 = s0[r] * SCALE_ + bcache[idxs[(size_t)row * N_ + c0]];
      float a1 = s1[r] * SCALE_ + bcache[idxs[(size_t)row * N_ + c0 + 16]];
      p0[r] = a0; p1[r] = a1;
      float t = fmaxf(a0, a1);
      t = fmaxf(t, __shfl_xor(t, 1));
      t = fmaxf(t, __shfl_xor(t, 2));
      t = fmaxf(t, __shfl_xor(t, 4));
      t = fmaxf(t, __shfl_xor(t, 8));
      rm[r] = t;
    }

    // online softmax update; stage P into per-wave LDS (16 rows x 32 cols f16)
    _Float16* Pw = Pl[wid];
#pragma unroll
    for (int r = 0; r < 8; ++r) {
      float mnew = fmaxf(mrun[r], rm[r]);
      float fs = __expf(mrun[r] - mnew);
      float e0 = __expf(p0[r] - mnew);
      float e1 = __expf(p1[r] - mnew);
      float rs = e0 + e1;
      rs += __shfl_xor(rs, 1);
      rs += __shfl_xor(rs, 2);
      rs += __shfl_xor(rs, 4);
      rs += __shfl_xor(rs, 8);
      lrun[r] = lrun[r] * fs + rs;
      mrun[r] = mnew;
#pragma unroll
      for (int dt = 0; dt < 8; ++dt) Oacc[dt][r] *= fs;
      int prow = r + half * 8;
      Pw[prow * 40 + l16] = (_Float16)e0;
      Pw[prow * 40 + 16 + l16] = (_Float16)e1;
    }

    // P as A fragment (same wave wrote it; DS ops in-order within wave)
    U16 pf;
    const unsigned int* P32 = (const unsigned int*)Pw;  // pitch 20 u32
    {
      const unsigned int* p = &P32[l16 * 20 + half * 4];
      pf.v4[0] = *(const uint4*)(p);
      pf.v4[1] = *(const uint4*)(p + 8);
    }

    // V fragments from vT (B,H,D,N layout => u32 {0..3}+half*4 and {8..11}+half*4), O += P V
#pragma unroll
    for (int dt = 0; dt < 8; ++dt) {
      U16 vf;
      const unsigned int* p =
          &v32[((size_t)bh * D_ + dt * 16 + l16) * 512 + (j0 >> 1) + half * 4];
      vf.v4[0] = *(const uint4*)(p);
      vf.v4[1] = *(const uint4*)(p + 8);
      if (dt == 0)
        Oacc[dt] = __builtin_amdgcn_wmma_f32_16x16x32_f16(false, pf.h, false, vf.h,
                                                          (short)0, Oacc[dt], false, false);
      else
        Oacc[dt] = __builtin_amdgcn_wmma_f32_16x16x32_f16(false, pf.h, false, vf.h,
                                                          (short)0, Oacc[dt], true, false);
    }
  }

  // normalize and store O as (B, N, H, D) f16 -> ready as GEMM A for proj
#pragma unroll
  for (int r = 0; r < 8; ++r) {
    float inv = 1.f / lrun[r];
    int row = q0 + r + half * 8;
#pragma unroll
    for (int dt = 0; dt < 8; ++dt) {
      int d = dt * 16 + l16;
      ob[(((size_t)b * N_ + row) * H_ + h) * D_ + d] = (_Float16)(Oacc[dt][r] * inv);
    }
  }
}

// ---------------------------------------------------------------------------
// launch
// ---------------------------------------------------------------------------
extern "C" void kernel_launch(void* const* d_in, const int* in_sizes, int n_in,
                              void* d_out, int out_size, void* d_ws, size_t ws_size,
                              hipStream_t stream) {
  const float* x     = (const float*)d_in[0];
  const float* lng   = (const float*)d_in[1];
  const float* lnb   = (const float*)d_in[2];
  const float* qkvw  = (const float*)d_in[3];
  const float* qkvb  = (const float*)d_in[4];
  const float* projw = (const float*)d_in[5];
  const float* projb = (const float*)d_in[6];
  const float* ab    = (const float*)d_in[7];
  const int*   idxs  = (const int*)d_in[8];
  float* out = (float*)d_out;

  // workspace layout (bytes, ~352 MB total)
  char* ws = (char*)d_ws;
  _Float16* xn     = (_Float16*)(ws);                          //  64 MB
  _Float16* qkvwT  = (_Float16*)(ws + ((size_t)64  << 20));    //  24 MB
  _Float16* projwT = (_Float16*)(ws + ((size_t)88  << 20));    //   8 MB
  _Float16* qbuf   = (_Float16*)(ws + ((size_t)96  << 20));    //  64 MB
  _Float16* kbuf   = (_Float16*)(ws + ((size_t)160 << 20));    //  64 MB
  _Float16* vtbuf  = (_Float16*)(ws + ((size_t)224 << 20));    //  64 MB
  _Float16* obuf   = (_Float16*)(ws + ((size_t)288 << 20));    //  64 MB

  ln_kernel<<<B_ * N_, 256, 0, stream>>>(x, lng, lnb, xn);
  wt_kernel<<<(DIM_ * QKVO_ + 255) / 256, 256, 0, stream>>>(qkvw, qkvwT, DIM_, QKVO_);
  wt_kernel<<<(DH_ * DIM_ + 255) / 256, 256, 0, stream>>>(projw, projwT, DH_, DIM_);

  gemm_kernel<<<dim3(QKVO_ / 64, (B_ * N_) / 128), 256, 0, stream>>>(
      xn, qkvwT, qkvb, DIM_, 0, qbuf, kbuf, vtbuf, nullptr);

  attn_kernel<<<B_ * H_ * (N_ / 64), 128, 0, stream>>>(
      qbuf, kbuf, vtbuf, ab, idxs, obuf);

  gemm_kernel<<<dim3(DIM_ / 64, (B_ * N_) / 128), 256, 0, stream>>>(
      obuf, projwT, projb, DH_, 1, nullptr, nullptr, nullptr, out);
}
